// CausalSelfAttention_23974507446868
// MI455X (gfx1250) — matrix operand order repaired and checked by reading, not compiled, not run on hardware
//
#include <hip/hip_runtime.h>

// ---------------------------------------------------------------------------
// CausalSelfAttention (GQA) for MI455X / gfx1250, wave32 + WMMA bf16.
//   B=4, T=2048, C=2048, HQ=16, HKV=4, HS=128
// Pipeline: cvt(x)->bf16 ; pack(W)->K-pair u32 ; GEMM1(WMMA, async-LDS
//           double buffered) -> qkv(bf16) ; flash-attention(WMMA) -> y(bf16) ;
//           GEMM2(WMMA)+bias -> out(f32)
// ---------------------------------------------------------------------------

#define DEV __device__ __forceinline__

typedef __attribute__((ext_vector_type(16))) __bf16 v16bf;
typedef __attribute__((ext_vector_type(8)))  float  v8f;

constexpr int Bb   = 4;
constexpr int T    = 2048;
constexpr int C    = 2048;
constexpr int HQ   = 16;
constexpr int HKV  = 4;
constexpr int HS   = 128;
constexpr int KVD  = HKV * HS;        // 512
constexpr int NQKV = C + 2 * KVD;     // 3072
constexpr int MROWS = Bb * T;         // 8192

// Fragment: 16 bf16 values = 8 VGPRs per lane (A or B operand of 16x16x32)
union Frag {
    v16bf v;
    uint4 q[2];
    unsigned int u[8];
};

DEV unsigned short f2bf(float f) {             // RNE f32 -> bf16 bits
    unsigned int u = __float_as_uint(f);
    unsigned int r = (u + 0x7FFFu + ((u >> 16) & 1u)) >> 16;
    return (unsigned short)r;
}

DEV v8f wmma_bf16(const Frag& a, const Frag& b, v8f c) {
    return __builtin_amdgcn_wmma_f32_16x16x32_bf16(
        false, a.v, false, b.v, (short)0, c, false, false);
}

// Async global -> LDS 16-byte copy (per active lane); tracked by ASYNCcnt.
// VDST operand = LDS byte address VGPR; generic-pointer low 32 bits are the
// DS offset (aperture truncation rule, ISA 10.2).
DEV void async_copy_b128(const void* gaddr, void* laddr) {
    asm volatile("global_load_async_to_lds_b128 %0, %1, off"
                 :: "v"((unsigned int)(unsigned long long)laddr),
                    "v"(gaddr)
                 : "memory");
}
DEV void wait_async0() {
    asm volatile("s_wait_asynccnt 0x0" ::: "memory");
}

// ---------------------------------------------------------------------------
// fp32 -> bf16 bulk convert (x)
// ---------------------------------------------------------------------------
__global__ void cvt_f32_to_bf16(const float* __restrict__ in,
                                unsigned short* __restrict__ out, int n4) {
    int i = blockIdx.x * blockDim.x + threadIdx.x;
    if (i >= n4) return;
    float4 f = ((const float4*)in)[i];
    ushort4 o;
    o.x = f2bf(f.x); o.y = f2bf(f.y); o.z = f2bf(f.z); o.w = f2bf(f.w);
    ((ushort4*)out)[i] = o;
}

// ---------------------------------------------------------------------------
// Pack weight fp32 [K][N] -> u32 [K/2][N], word = {lo: bf16 W[2k][n],
// hi: bf16 W[2k+1][n]}  == the 16-bit B-matrix K-pair VGPR layout.
// ---------------------------------------------------------------------------
__global__ void pack_w_bf16(const float* __restrict__ w,
                            unsigned int* __restrict__ wp, int K, int N) {
    int idx = blockIdx.x * blockDim.x + threadIdx.x;
    int total = (K >> 1) * N;
    if (idx >= total) return;
    int k2 = idx / N, n = idx - k2 * N;
    unsigned int lo = f2bf(w[(size_t)(2 * k2) * N + n]);
    unsigned int hi = f2bf(w[(size_t)(2 * k2 + 1) * N + n]);
    wp[idx] = lo | (hi << 16);
}

// ---------------------------------------------------------------------------
// bf16 WMMA GEMM:  D[M][N] = A[M][K](bf16) * Bpacked[K/2][N] + bias[N]
// Block: 256 threads = 8 wave32, tile 128x128, K-step 32.
// Double-buffered LDS staged by GLOBAL_LOAD_ASYNC_TO_LDS_B128 so the DMA of
// tile k+1 overlaps WMMA on tile k.  One barrier per K-step.
// ---------------------------------------------------------------------------
template <bool STORE_BF16>
__global__ __launch_bounds__(256)
void gemm_bf16_wmma(const unsigned short* __restrict__ A,
                    const unsigned int*  __restrict__ Bp,
                    const float*         __restrict__ bias,
                    unsigned short*      __restrict__ Dbf,
                    float*               __restrict__ Df,
                    int M, int N, int K)
{
    __shared__ alignas(16) unsigned short ldsA[2][128 * 32];  // 2 x 8 KB
    __shared__ alignas(16) unsigned int   ldsB[2][16 * 128];  // 2 x 8 KB

    const int tid  = threadIdx.x;
    const int lane = tid & 31;
    const int wave = tid >> 5;
    const int h    = lane >> 4;       // half of wave (0/1)
    const int ln   = lane & 15;
    const int wm   = wave >> 2;       // 0..1
    const int wn   = wave & 3;        // 0..3
    const int m0   = blockIdx.y * 128;
    const int n0   = blockIdx.x * 128;

    // per-thread staging coordinates (2 x 16B chunks for A and for B)
    const int ra0 = tid >> 2,          oa0 = (tid & 3) * 8;
    const int ra1 = (tid + 256) >> 2,  oa1 = ((tid + 256) & 3) * 8;
    const int rb0 = tid >> 5,          cb0 = (tid & 31) * 4;
    const int rb1 = (tid + 256) >> 5,  cb1 = ((tid + 256) & 31) * 4;

    v8f acc[4][2];
    #pragma unroll
    for (int i = 0; i < 4; ++i)
        #pragma unroll
        for (int j = 0; j < 2; ++j) acc[i][j] = {};

    // stage tile 0 into buffer 0
    {
        async_copy_b128(&A[(size_t)(m0 + ra0) * K + oa0],            &ldsA[0][ra0 * 32 + oa0]);
        async_copy_b128(&A[(size_t)(m0 + ra1) * K + oa1],            &ldsA[0][ra1 * 32 + oa1]);
        async_copy_b128(&Bp[(size_t)rb0 * N + n0 + cb0],             &ldsB[0][rb0 * 128 + cb0]);
        async_copy_b128(&Bp[(size_t)rb1 * N + n0 + cb1],             &ldsB[0][rb1 * 128 + cb1]);
    }

    int buf = 0;
    for (int k0 = 0; k0 < K; k0 += 32) {
        wait_async0();          // my async copies for `buf` have landed
        __syncthreads();        // everyone's have; prior readers of buf^1 done

        if (k0 + 32 < K) {      // kick DMA for the next tile into buf^1
            int kn = k0 + 32;
            async_copy_b128(&A[(size_t)(m0 + ra0) * K + kn + oa0],        &ldsA[buf ^ 1][ra0 * 32 + oa0]);
            async_copy_b128(&A[(size_t)(m0 + ra1) * K + kn + oa1],        &ldsA[buf ^ 1][ra1 * 32 + oa1]);
            async_copy_b128(&Bp[(size_t)((kn >> 1) + rb0) * N + n0 + cb0], &ldsB[buf ^ 1][rb0 * 128 + cb0]);
            async_copy_b128(&Bp[(size_t)((kn >> 1) + rb1) * N + n0 + cb1], &ldsB[buf ^ 1][rb1 * 128 + cb1]);
        }

        // A fragments: lane<16 -> K 0..15 ; lane>=16 -> 16..31
        Frag af[4];
        #pragma unroll
        for (int fm = 0; fm < 4; ++fm) {
            int row = wm * 64 + fm * 16 + ln;
            af[fm].q[0] = *(const uint4*)&ldsA[buf][row * 32 + h * 16];
            af[fm].q[1] = *(const uint4*)&ldsA[buf][row * 32 + h * 16 + 8];
        }
        // B fragments: VGPR j of lanes(h) holds packed K-pair row (h*8+j)
        Frag bf[2];
        #pragma unroll
        for (int fn = 0; fn < 2; ++fn) {
            int col = wn * 32 + fn * 16 + ln;
            #pragma unroll
            for (int j = 0; j < 8; ++j)
                bf[fn].u[j] = ldsB[buf][(h * 8 + j) * 128 + col];
        }
        #pragma unroll
        for (int fm = 0; fm < 4; ++fm)
            #pragma unroll
            for (int fn = 0; fn < 2; ++fn)
                acc[fm][fn] = wmma_bf16(af[fm], bf[fn], acc[fm][fn]);

        buf ^= 1;
    }

    // epilogue: C layout VGPR j -> row M = 8*h + j, col N = ln
    #pragma unroll
    for (int fn = 0; fn < 2; ++fn) {
        int col = n0 + wn * 32 + fn * 16 + ln;
        float bv = bias[col];
        #pragma unroll
        for (int fm = 0; fm < 4; ++fm) {
            #pragma unroll
            for (int j = 0; j < 8; ++j) {
                int row = m0 + wm * 64 + fm * 16 + h * 8 + j;
                float v = acc[fm][fn][j] + bv;
                if (STORE_BF16) Dbf[(size_t)row * N + col] = f2bf(v);
                else            Df [(size_t)row * N + col] = v;
            }
        }
    }
}

// ---------------------------------------------------------------------------
// Flash-attention with WMMA.  Block = 4 wave32, one (b, h), 64 queries
// (16 per wave).  Key tiles of 32 streamed with online softmax.
// ---------------------------------------------------------------------------
__global__ __launch_bounds__(128)
void attn_wmma(const unsigned short* __restrict__ qkv,  // [B*T][3072] bf16
               unsigned short* __restrict__ yb)          // [B*T][2048] bf16
{
    __shared__ alignas(16) unsigned int   ldsV[16 * 128];     // packed V 32x128
    __shared__ alignas(16) unsigned short ldsP[4][16 * 32];   // per-wave P

    const int tid  = threadIdx.x;
    const int lane = tid & 31;
    const int wave = tid >> 5;
    const int h    = lane >> 4;
    const int ln   = lane & 15;

    const int bid  = blockIdx.x;
    const int qblk = bid & 31;          // T/64 = 32 query blocks
    const int bh   = bid >> 5;
    const int b    = bh >> 4;
    const int hq   = bh & 15;
    const int hk   = hq >> 2;           // GQA: rep = 4

    const int qbase = qblk * 64 + wave * 16;

    const unsigned short* qhead = qkv + (size_t)(b * T) * NQKV + hq * HS;
    const unsigned short* khead = qkv + (size_t)(b * T) * NQKV + C + hk * HS;
    const unsigned short* vhead = qkv + (size_t)(b * T) * NQKV + C + KVD + hk * HS;

    // Preload Q A-fragments for the 4 K-steps over d=128.
    Frag qf[4];
    {
        const unsigned short* qp = qhead + (size_t)(qbase + ln) * NQKV;
        #pragma unroll
        for (int s = 0; s < 4; ++s) {
            qf[s].q[0] = *(const uint4*)&qp[s * 32 + h * 16];
            qf[s].q[1] = *(const uint4*)&qp[s * 32 + h * 16 + 8];
        }
    }

    v8f o[8];
    #pragma unroll
    for (int i = 0; i < 8; ++i) o[i] = {};
    float mrow[8], lrow[8];
    #pragma unroll
    for (int i = 0; i < 8; ++i) { mrow[i] = -1e30f; lrow[i] = 0.f; }

    const float scale = 0.08838834764831845f;   // 1/sqrt(128)
    const int nkt = qblk * 2 + 2;               // causal key-tile count

    for (int kt = 0; kt < nkt; ++kt) {
        const int kbase = kt * 32;

        // ---- cooperative repack of V tile (32x128) into K-pair LDS layout
        {
            int k2 = tid & 15;
            int g  = tid >> 4;  // 0..7, 16 d-columns each
            const unsigned int* v0 =
                (const unsigned int*)(vhead + (size_t)(kbase + 2 * k2) * NQKV);
            const unsigned int* v1 =
                (const unsigned int*)(vhead + (size_t)(kbase + 2 * k2 + 1) * NQKV);
            #pragma unroll
            for (int i = 0; i < 8; ++i) {
                unsigned int a = v0[g * 8 + i];
                unsigned int c = v1[g * 8 + i];
                ldsV[k2 * 128 + 2 * (g * 8 + i) + 0] = (a & 0xFFFFu) | (c << 16);
                ldsV[k2 * 128 + 2 * (g * 8 + i) + 1] = (a >> 16) | (c & 0xFFFF0000u);
            }
        }
        __syncthreads();

        if (kbase <= qbase + 15) {          // wave-uniform: EXEC stays full
            // ---- S = (q @ k^T): k rows feed B-fragments directly
            v8f s0 = {}, s1 = {};
            const unsigned int* kp0 =
                (const unsigned int*)(khead + (size_t)(kbase + ln) * NQKV);
            const unsigned int* kp1 =
                (const unsigned int*)(khead + (size_t)(kbase + 16 + ln) * NQKV);
            #pragma unroll
            for (int s = 0; s < 4; ++s) {
                Frag kf0, kf1;
                kf0.q[0] = *(const uint4*)&kp0[s * 16 + h * 8];
                kf0.q[1] = *(const uint4*)&kp0[s * 16 + h * 8 + 4];
                kf1.q[0] = *(const uint4*)&kp1[s * 16 + h * 8];
                kf1.q[1] = *(const uint4*)&kp1[s * 16 + h * 8 + 4];
                s0 = wmma_bf16(qf[s], kf0, s0);
                s1 = wmma_bf16(qf[s], kf1, s1);
            }

            // ---- scale + causal mask (row q = qbase + 8h + j, key = kbase+ln)
            #pragma unroll
            for (int j = 0; j < 8; ++j) {
                int q = qbase + h * 8 + j;
                float a0 = s0[j] * scale;
                float a1 = s1[j] * scale;
                if (kbase + ln > q)      a0 = -1e30f;
                if (kbase + 16 + ln > q) a1 = -1e30f;
                s0[j] = a0; s1[j] = a1;
            }

            // ---- online softmax (row = 16 lanes of one half, butterfly)
            float alpha[8];
            unsigned short* Prow = ldsP[wave];
            #pragma unroll
            for (int j = 0; j < 8; ++j) {
                float mt = fmaxf(s0[j], s1[j]);
                #pragma unroll
                for (int d = 8; d >= 1; d >>= 1)
                    mt = fmaxf(mt, __shfl_xor(mt, d, 32));
                float mnew = fmaxf(mrow[j], mt);
                float al = __expf(mrow[j] - mnew);
                float p0 = __expf(s0[j] - mnew);
                float p1 = __expf(s1[j] - mnew);
                float rs = p0 + p1;
                #pragma unroll
                for (int d = 8; d >= 1; d >>= 1)
                    rs += __shfl_xor(rs, d, 32);
                mrow[j] = mnew;
                lrow[j] = lrow[j] * al + rs;
                alpha[j] = al;
                // P (C-layout) -> bf16 row-major scratch for A-fragment reload
                Prow[(h * 8 + j) * 32 + ln]      = f2bf(p0);
                Prow[(h * 8 + j) * 32 + 16 + ln] = f2bf(p1);
            }
            #pragma unroll
            for (int nt = 0; nt < 8; ++nt)
                #pragma unroll
                for (int j = 0; j < 8; ++j)
                    o[nt][j] *= alpha[j];

            asm volatile("s_wait_dscnt 0" ::: "memory");   // P stores visible

            Frag pf;   // A-fragment of P (16x32)
            pf.q[0] = *(const uint4*)&Prow[ln * 32 + h * 16];
            pf.q[1] = *(const uint4*)&Prow[ln * 32 + h * 16 + 8];

            // ---- O += P @ V : 8 column tiles of 16
            #pragma unroll
            for (int nt = 0; nt < 8; ++nt) {
                Frag vf;
                #pragma unroll
                for (int j = 0; j < 8; ++j)
                    vf.u[j] = ldsV[(h * 8 + j) * 128 + nt * 16 + ln];
                o[nt] = wmma_bf16(pf, vf, o[nt]);
            }
        }
        __syncthreads();
    }

    // ---- normalize and store y (bf16) for the projection GEMM
    #pragma unroll
    for (int j = 0; j < 8; ++j) {
        float inv = 1.0f / lrow[j];
        int qrow = qbase + h * 8 + j;
        unsigned short* yrow = yb + (size_t)(b * T + qrow) * C + hq * HS;
        #pragma unroll
        for (int nt = 0; nt < 8; ++nt)
            yrow[nt * 16 + ln] = f2bf(o[nt][j] * inv);
    }
}

// ---------------------------------------------------------------------------
// Launch
// ---------------------------------------------------------------------------
extern "C" void kernel_launch(void* const* d_in, const int* in_sizes, int n_in,
                              void* d_out, int out_size, void* d_ws, size_t ws_size,
                              hipStream_t stream) {
    const float* x     = (const float*)d_in[0];
    const float* Wattn = (const float*)d_in[1];
    const float* battn = (const float*)d_in[2];
    const float* Wproj = (const float*)d_in[3];
    const float* bproj = (const float*)d_in[4];
    float* out = (float*)d_out;

    // workspace layout (bytes); yb aliases xb (x dead after GEMM1)
    char* ws = (char*)d_ws;
    unsigned short* xb      = (unsigned short*)(ws);               // 33,554,432
    unsigned int*   wattn_p = (unsigned int*)(ws + 33554432);      // 12,582,912
    unsigned int*   wproj_p = (unsigned int*)(ws + 46137344);      //  8,388,608
    unsigned short* qkvb    = (unsigned short*)(ws + 54525952);    // 50,331,648
    unsigned short* yb      = xb;                                  // reuse

    {
        int n4 = (Bb * T * C) / 4;
        cvt_f32_to_bf16<<<(n4 + 255) / 256, 256, 0, stream>>>(x, xb, n4);
    }
    {
        int t1 = (C / 2) * NQKV;
        pack_w_bf16<<<(t1 + 255) / 256, 256, 0, stream>>>(Wattn, wattn_p, C, NQKV);
        int t2 = (C / 2) * C;
        pack_w_bf16<<<(t2 + 255) / 256, 256, 0, stream>>>(Wproj, wproj_p, C, C);
    }
    {   // qkv = x @ W_attn + b_attn  -> bf16
        dim3 grid(NQKV / 128, MROWS / 128);
        gemm_bf16_wmma<true><<<grid, 256, 0, stream>>>(
            xb, wattn_p, battn, qkvb, nullptr, MROWS, NQKV, C);
    }
    {   // flash attention -> y bf16
        attn_wmma<<<Bb * HQ * (T / 64), 128, 0, stream>>>(qkvb, yb);
    }
    {   // out = y @ W_proj + b_proj -> f32
        dim3 grid(C / 128, MROWS / 128);
        gemm_bf16_wmma<false><<<grid, 256, 0, stream>>>(
            yb, wproj_p, bproj, nullptr, out, MROWS, C, C);
    }
}
